// GNN_33285996544495
// MI455X (gfx1250) — matrix operand order repaired and checked by reading, not compiled, not run on hardware
//
#include <hip/hip_runtime.h>

#define N_NODES 100000
#define N_EDGES 1600000
#define FEAT    64
#define NGRAPH  64
#define ALPHA   0.5f

typedef __attribute__((ext_vector_type(16))) __bf16 v16bf;
typedef __attribute__((ext_vector_type(8)))  float  v8f;

union FragU { uint4 u[2]; v16bf v; };

// float -> bf16 with round-to-nearest-even (bit twiddle; used only on the A side)
static __device__ __forceinline__ __bf16 f2bf(float f) {
  unsigned u = __float_as_uint(f);
  unsigned r = u + 0x7FFFu + ((u >> 16) & 1u);
  unsigned short h = (unsigned short)(r >> 16);
  __bf16 b;
  __builtin_memcpy(&b, &h, 2);
  return b;
}

// ---------------- zero init ----------------
__global__ void zero_f32(float* __restrict__ p, int n) {
  int i = blockIdx.x * blockDim.x + threadIdx.x;
  if (i < n) p[i] = 0.0f;
}

// ---------------- degrees ----------------
__global__ void degree_kernel(const int* __restrict__ row, const int* __restrict__ col,
                              float* __restrict__ deg_out, float* __restrict__ deg_in) {
  int e = blockIdx.x * blockDim.x + threadIdx.x;
  if (e >= N_EDGES) return;
  atomicAdd(deg_out + row[e], 1.0f);
  atomicAdd(deg_in  + col[e], 1.0f);
}

// deg -> deg^{-1/2} (0 if deg==0), applied to [r_out | r_in] contiguously (2N)
__global__ void invsqrt_kernel(float* __restrict__ d, int n) {
  int i = blockIdx.x * blockDim.x + threadIdx.x;
  if (i >= n) return;
  float v = d[i];
  d[i] = (v > 0.0f) ? rsqrtf(v) : 0.0f;
}

__global__ void norm_kernel(const int* __restrict__ row, const int* __restrict__ col,
                            const float* __restrict__ r_out, const float* __restrict__ r_in,
                            float* __restrict__ norm) {
  int e = blockIdx.x * blockDim.x + threadIdx.x;
  if (e >= N_EDGES) return;
  norm[e] = r_out[row[e]] * r_in[col[e]];
}

// ---------------- edge scatter: aggA[row] += n*x[col], aggT[col] += n*x[row] --------
// one thread = (edge, 4-feature chunk); 16 threads per edge
__global__ void scatter_kernel(const int* __restrict__ row, const int* __restrict__ col,
                               const float* __restrict__ norm, const float* __restrict__ x,
                               float* __restrict__ aggA, float* __restrict__ aggT) {
  int gid = blockIdx.x * blockDim.x + threadIdx.x;
  if (gid >= N_EDGES * 16) return;
  int e = gid >> 4;
  int f = (gid & 15) << 2;
  int r = row[e];
  int c = col[e];
  float w = norm[e];
  float4 xs = *(const float4*)(x + c * FEAT + f);  // -> aggA[row]
  float4 xd = *(const float4*)(x + r * FEAT + f);  // -> aggT[col]
  float* pa = aggA + r * FEAT + f;
  float* pt = aggT + c * FEAT + f;
  atomicAdd(pa + 0, w * xs.x);
  atomicAdd(pa + 1, w * xs.y);
  atomicAdd(pa + 2, w * xs.z);
  atomicAdd(pa + 3, w * xs.w);
  atomicAdd(pt + 0, w * xd.x);
  atomicAdd(pt + 1, w * xd.y);
  atomicAdd(pt + 2, w * xd.z);
  atomicAdd(pt + 3, w * xd.w);
}

// ---- pre-pack one 64x64 weight matrix into WMMA B-fragment lane order (bf16) ----
// 256 threads = 8 fragments (nt 0..3 x kt 0..1) x 32 lanes; each thread emits its
// 16-element lane fragment contiguously (32 bytes) so the GEMM loads it as 2x b128.
__global__ void pack_w_bf16(const float* __restrict__ w, unsigned short* __restrict__ o) {
  int t    = threadIdx.x;    // 0..255
  int lane = t & 31;
  int frag = t >> 5;         // 0..7 = nt*2+kt
  int nt   = frag >> 1;
  int kt   = frag & 1;
  int hl   = lane >> 4;
  int l16  = lane & 15;
  int col  = nt * 16 + l16;
  unsigned short* dst = o + (frag * 32 + lane) * 16;
#pragma unroll
  for (int e = 0; e < 16; ++e) {
    int k0 = kt * 32 + hl * 8 + (e & 7) + (e >> 3) * 16;   // B 32x16: lanes 0-15 K=0..7,16..23, lanes 16-31 K=8..15,24..31
    unsigned u = __float_as_uint(w[k0 * FEAT + col]);
    unsigned r = u + 0x7FFFu + ((u >> 16) & 1u);
    dst[e] = (unsigned short)(r >> 16);
  }
}

static __device__ __forceinline__ v16bf load_frag(const uint4* __restrict__ p, int frag, int lane) {
  FragU f;
  f.u[0] = p[(frag * 32 + lane) * 2 + 0];
  f.u[1] = p[(frag * 32 + lane) * 2 + 1];
  return f.v;
}

// -------- fused dual-GEMM + bias + relu:  out = relu(a*aggA@Ws + (1-a)*aggT@Wd + bias) -----
// One wave per 16-node tile; 10 waves (320 threads) per block; 6250 tiles = 625 blocks exactly.
__global__ __launch_bounds__(320)
void gcn_gemm_relu(const float* __restrict__ aggA, const float* __restrict__ aggT,
                   const uint4* __restrict__ packS, const uint4* __restrict__ packD,
                   const float* __restrict__ bS, const float* __restrict__ bD,
                   float* __restrict__ out) {
  int wave = threadIdx.x >> 5;
  int lane = threadIdx.x & 31;
  int hl   = lane >> 4;
  int l16  = lane & 15;
  int tile = blockIdx.x * 10 + wave;   // 0..6249, exact (100000 = 6250*16)
  int m0   = tile * 16;
  int mrow = m0 + l16;

  // A fragments (16x32 bf16): lane element e -> K = kt*32 + hl*8 + (e&7) + (e>>3)*16
  // i.e. two contiguous 8-float runs per kt -> four float4 loads each.
  const float4* pA4 = (const float4*)(aggA + mrow * FEAT);
  const float4* pT4 = (const float4*)(aggT + mrow * FEAT);
  int b = hl * 2;
  float4 sa[8], sd[8];
  sa[0] = pA4[b + 0];  sa[1] = pA4[b + 1];  sa[2] = pA4[b + 4];  sa[3] = pA4[b + 5];
  sa[4] = pA4[b + 8];  sa[5] = pA4[b + 9];  sa[6] = pA4[b + 12]; sa[7] = pA4[b + 13];
  sd[0] = pT4[b + 0];  sd[1] = pT4[b + 1];  sd[2] = pT4[b + 4];  sd[3] = pT4[b + 5];
  sd[4] = pT4[b + 8];  sd[5] = pT4[b + 9];  sd[6] = pT4[b + 12]; sd[7] = pT4[b + 13];

  v16bf as0, as1, ad0, ad1;
#pragma unroll
  for (int q = 0; q < 4; ++q) {
    float4 v0 = sa[q], v1 = sa[q + 4], t0 = sd[q], t1 = sd[q + 4];
    as0[q * 4 + 0] = f2bf(ALPHA * v0.x);  as0[q * 4 + 1] = f2bf(ALPHA * v0.y);
    as0[q * 4 + 2] = f2bf(ALPHA * v0.z);  as0[q * 4 + 3] = f2bf(ALPHA * v0.w);
    as1[q * 4 + 0] = f2bf(ALPHA * v1.x);  as1[q * 4 + 1] = f2bf(ALPHA * v1.y);
    as1[q * 4 + 2] = f2bf(ALPHA * v1.z);  as1[q * 4 + 3] = f2bf(ALPHA * v1.w);
    ad0[q * 4 + 0] = f2bf((1.0f - ALPHA) * t0.x);  ad0[q * 4 + 1] = f2bf((1.0f - ALPHA) * t0.y);
    ad0[q * 4 + 2] = f2bf((1.0f - ALPHA) * t0.z);  ad0[q * 4 + 3] = f2bf((1.0f - ALPHA) * t0.w);
    ad1[q * 4 + 0] = f2bf((1.0f - ALPHA) * t1.x);  ad1[q * 4 + 1] = f2bf((1.0f - ALPHA) * t1.y);
    ad1[q * 4 + 2] = f2bf((1.0f - ALPHA) * t1.z);  ad1[q * 4 + 3] = f2bf((1.0f - ALPHA) * t1.w);
  }

#pragma unroll
  for (int nt = 0; nt < 4; ++nt) {
    int colp = nt * 16 + l16;
    v16bf fs0 = load_frag(packS, nt * 2 + 0, lane);
    v16bf fs1 = load_frag(packS, nt * 2 + 1, lane);
    v16bf fd0 = load_frag(packD, nt * 2 + 0, lane);
    v16bf fd1 = load_frag(packD, nt * 2 + 1, lane);
    float biasv = ALPHA * bS[colp] + (1.0f - ALPHA) * bD[colp];
    v8f acc;
#pragma unroll
    for (int v = 0; v < 8; ++v) acc[v] = biasv;
    acc = __builtin_amdgcn_wmma_f32_16x16x32_bf16(false, as0, false, fs0, (short)0, acc, false, false);
    acc = __builtin_amdgcn_wmma_f32_16x16x32_bf16(false, as1, false, fs1, (short)0, acc, false, false);
    acc = __builtin_amdgcn_wmma_f32_16x16x32_bf16(false, ad0, false, fd0, (short)0, acc, false, false);
    acc = __builtin_amdgcn_wmma_f32_16x16x32_bf16(false, ad1, false, fd1, (short)0, acc, false, false);
    // C/D layout: VGPR v, lanes 0-15 -> M=v, lanes 16-31 -> M=v+8
#pragma unroll
    for (int v = 0; v < 8; ++v) {
      int m = m0 + hl * 8 + v;
      float r = acc[v];
      out[m * FEAT + colp] = r > 0.0f ? r : 0.0f;
    }
  }
}

// ---------------- global max pool (h >= 0 after relu -> uint-bit atomicMax) ------------
__global__ void pool_max(const float* __restrict__ h, const int* __restrict__ batch,
                         unsigned* __restrict__ g) {
  int i = blockIdx.x * blockDim.x + threadIdx.x;
  if (i >= N_NODES * FEAT) return;
  int n = i >> 6;
  int f = i & 63;
  atomicMax(g + batch[n] * FEAT + f, __float_as_uint(h[i]));
}

// ---------------- tiny MLP head ----------------
__global__ void head_kernel(const float* __restrict__ g,
                            const float* __restrict__ wl1, const float* __restrict__ bl1,
                            const float* __restrict__ wl2, const float* __restrict__ bl2,
                            float* __restrict__ out) {
  int i = threadIdx.x;
  if (i >= NGRAPH) return;
  const float* gi = g + i * FEAT;
  float o = bl2[0];
#pragma unroll
  for (int j = 0; j < 5; ++j) {
    float s = bl1[j];
#pragma unroll 8
    for (int k = 0; k < FEAT; ++k) s += gi[k] * wl1[k * 5 + j];
    s = s > 0.0f ? s : 0.0f;
    o += s * wl2[j];
  }
  out[i] = o;
}

extern "C" void kernel_launch(void* const* d_in, const int* in_sizes, int n_in,
                              void* d_out, int out_size, void* d_ws, size_t ws_size,
                              hipStream_t stream) {
  const float* x     = (const float*)d_in[0];
  const int*   ei    = (const int*)  d_in[1];
  const int*   row   = ei;
  const int*   col   = ei + N_EDGES;
  const int*   batch = (const int*)  d_in[2];
  const float* w1s = (const float*)d_in[3];  const float* b1s = (const float*)d_in[4];
  const float* w1d = (const float*)d_in[5];  const float* b1d = (const float*)d_in[6];
  const float* w2s = (const float*)d_in[7];  const float* b2s = (const float*)d_in[8];
  const float* w2d = (const float*)d_in[9];  const float* b2d = (const float*)d_in[10];
  const float* w3s = (const float*)d_in[11]; const float* b3s = (const float*)d_in[12];
  const float* w3d = (const float*)d_in[13]; const float* b3d = (const float*)d_in[14];
  const float* wl1 = (const float*)d_in[15]; const float* bl1 = (const float*)d_in[16];
  const float* wl2 = (const float*)d_in[17]; const float* bl2 = (const float*)d_in[18];

  // workspace layout (floats)
  float* wsf   = (float*)d_ws;
  float* r_out = wsf;                       // N
  float* r_in  = r_out + N_NODES;           // N
  float* nrm   = r_in + N_NODES;            // E
  float* aggA  = nrm + N_EDGES;             // N*64
  float* aggT  = aggA + N_NODES * FEAT;     // N*64
  float* h0    = aggT + N_NODES * FEAT;     // N*64
  float* h1    = h0 + N_NODES * FEAT;       // N*64
  float* gbuf  = h1 + N_NODES * FEAT;       // 64*64
  // packed bf16 B-fragments (4096 bf16 = 2048 floats each); base is 32B aligned
  unsigned short* pk  = (unsigned short*)(gbuf + NGRAPH * FEAT);
  unsigned short* p1s = pk;
  unsigned short* p1d = p1s + 4096;
  unsigned short* p2s = p1d + 4096;
  unsigned short* p2d = p2s + 4096;
  unsigned short* p3s = p2d + 4096;
  unsigned short* p3d = p3s + 4096;

  const int B = 256;

  // pre-pack all six weight matrices into WMMA B-fragment order
  pack_w_bf16<<<1, 256, 0, stream>>>(w1s, p1s);
  pack_w_bf16<<<1, 256, 0, stream>>>(w1d, p1d);
  pack_w_bf16<<<1, 256, 0, stream>>>(w2s, p2s);
  pack_w_bf16<<<1, 256, 0, stream>>>(w2d, p2d);
  pack_w_bf16<<<1, 256, 0, stream>>>(w3s, p3s);
  pack_w_bf16<<<1, 256, 0, stream>>>(w3d, p3d);

  // degrees + norms (shared by all 3 layers)
  zero_f32<<<(2 * N_NODES + B - 1) / B, B, 0, stream>>>(r_out, 2 * N_NODES);
  degree_kernel<<<(N_EDGES + B - 1) / B, B, 0, stream>>>(row, col, r_out, r_in);
  invsqrt_kernel<<<(2 * N_NODES + B - 1) / B, B, 0, stream>>>(r_out, 2 * N_NODES);
  norm_kernel<<<(N_EDGES + B - 1) / B, B, 0, stream>>>(row, col, r_out, r_in, nrm);

  const int SCATTER_T = N_EDGES * 16;
  const int AGG_T = 2 * N_NODES * FEAT;

  // layer 1: x -> h0
  zero_f32<<<(AGG_T + B - 1) / B, B, 0, stream>>>(aggA, AGG_T);
  scatter_kernel<<<(SCATTER_T + B - 1) / B, B, 0, stream>>>(row, col, nrm, x, aggA, aggT);
  gcn_gemm_relu<<<625, 320, 0, stream>>>(aggA, aggT, (const uint4*)p1s, (const uint4*)p1d, b1s, b1d, h0);

  // layer 2: h0 -> h1
  zero_f32<<<(AGG_T + B - 1) / B, B, 0, stream>>>(aggA, AGG_T);
  scatter_kernel<<<(SCATTER_T + B - 1) / B, B, 0, stream>>>(row, col, nrm, h0, aggA, aggT);
  gcn_gemm_relu<<<625, 320, 0, stream>>>(aggA, aggT, (const uint4*)p2s, (const uint4*)p2d, b2s, b2d, h1);

  // layer 3: h1 -> h0
  zero_f32<<<(AGG_T + B - 1) / B, B, 0, stream>>>(aggA, AGG_T);
  scatter_kernel<<<(SCATTER_T + B - 1) / B, B, 0, stream>>>(row, col, nrm, h1, aggA, aggT);
  gcn_gemm_relu<<<625, 320, 0, stream>>>(aggA, aggT, (const uint4*)p3s, (const uint4*)p3d, b3s, b3d, h0);

  // pool + head
  zero_f32<<<(NGRAPH * FEAT + B - 1) / B, B, 0, stream>>>(gbuf, NGRAPH * FEAT);
  pool_max<<<(N_NODES * FEAT + B - 1) / B, B, 0, stream>>>(h0, batch, (unsigned*)gbuf);
  head_kernel<<<1, 64, 0, stream>>>(gbuf, wl1, bl1, wl2, bl2, (float*)d_out);
}
